// Model_19310172963148
// MI455X (gfx1250) — matrix operand order repaired and checked by reading, not compiled, not run on hardware
//
#include <hip/hip_runtime.h>
#include <hip/hip_bf16.h>
#include <math.h>

// ---------------------------------------------------------------------------
// VQ-VAE forward for MI455X (gfx1250, wave32, WMMA + TDM).
// Convs (incl. transposed) = implicit GEMM via v_wmma_f32_16x16x32_f16 with
// 4-way register blocking. Weight tiles are DMA'd into LDS by the Tensor Data
// Mover (tensor_load_to_lds, TENSORcnt), with hardware OOB zero-fill and LDS
// pad fields reproducing the 40-half bank-spread stride. im2col B tiles are
// gathered by VALU concurrently with the TDM transfer.
// ---------------------------------------------------------------------------

typedef __attribute__((ext_vector_type(16))) _Float16 v16h;
typedef __attribute__((ext_vector_type(8)))  _Float16 v8h;
typedef __attribute__((ext_vector_type(8)))  float    v8f;

#if __has_builtin(__builtin_amdgcn_tensor_load_to_lds) && \
    __has_builtin(__builtin_amdgcn_s_wait_tensorcnt)
#define USE_TDM 1
typedef unsigned int u32x4 __attribute__((ext_vector_type(4)));
typedef int          i32x8 __attribute__((ext_vector_type(8)));
typedef int          i32x4 __attribute__((ext_vector_type(4)));
#else
#define USE_TDM 0
#endif

#define LDSW 40   // padded row/col stride in halfs (80B) to spread LDS banks

// Block = 4 waves. Tile = (MW*16) M  x  ((4/MW)*64) N.  K-step = 32.
// Each wave: 16 M rows x 64 N cols = 4 accumulators, A fragment reused 4x.
// M = Cout, N = B*OH*OW, K = Cin*KH*KW. W is pre-packed f16 (Cout x Ktot).
template<int KH,int KW,int S,int P,int PRE,int POST,int RES,int TR,int VQ,int MW>
__global__ __launch_bounds__(128)
void conv_wmma(const float* __restrict__ X, const _Float16* __restrict__ W,
               const float* __restrict__ bias, const float* __restrict__ resid,
               const float* __restrict__ aux, float* __restrict__ out,
               int Bn, int Cin, int Cout, int Hin, int Win, int Hout, int Wout)
{
    constexpr int NW    = 4 / MW;       // wave groups along N
    constexpr int MTILE = MW * 16;
    constexpr int NCOLS = NW * 64;

    const int Ktot = Cin * KH * KW;
    const int Ntot = Bn * Hout * Wout;
    const int m0 = blockIdx.y * MTILE;
    const int n0 = blockIdx.x * NCOLS;
    const int tid  = threadIdx.x;
    const int wave = tid >> 5;
    const int lane = tid & 31;

    __shared__ _Float16 sA[MTILE * LDSW];   // A tile: MTILE rows x 32 K (row-major)
    __shared__ _Float16 sB[NCOLS * LDSW];   // B tile: NCOLS cols x 32 K (col-major)

#if USE_TDM
    const unsigned ldsA = (unsigned)(unsigned long long)&sA[0];
    const unsigned long long wbase =
        (unsigned long long)(size_t)W + (unsigned long long)m0 * (unsigned)Ktot * 2ull;
#endif

    // ---- fixed B-column ownership for this thread, pixel decomp hoisted ----
    int  myCol[2];  int nCols;  int kkBeg, kkEnd;
    if (MW == 4)      { myCol[0] = tid & 63;  myCol[1] = 0;        nCols = 1;
                        kkBeg = (tid >> 6) * 16; kkEnd = kkBeg + 16; }
    else if (MW == 2) { myCol[0] = tid;       myCol[1] = 0;        nCols = 1;
                        kkBeg = 0; kkEnd = 32; }
    else              { myCol[0] = tid;       myCol[1] = tid + 128; nCols = 2;
                        kkBeg = 0; kkEnd = 32; }

    int  cok[2];  long cbase[2];  int cy0[2], cx0[2];
    #pragma unroll
    for (int c = 0; c < 2; ++c) {
        cok[c] = 0; cbase[c] = 0; cy0[c] = 0; cx0[c] = 0;
        if (c < nCols) {
            const int n = n0 + myCol[c];
            if (n < Ntot) {
                const int ox = n % Wout; int t = n / Wout;
                const int oy = t % Hout; const int b = t / Hout;
                cok[c]   = 1;
                cbase[c] = (long)b * Cin * Hin * Win;
                if (TR) { cy0[c] = oy + P;     cx0[c] = ox + P; }
                else    { cy0[c] = oy * S - P; cx0[c] = ox * S - P; }
            }
        }
    }

    v8f acc[4];
    #pragma unroll
    for (int r = 0; r < 4; ++r) acc[r] = (v8f){};

    for (int k0 = 0; k0 < Ktot; k0 += 32) {
        __syncthreads();    // previous iteration's fragment reads complete
#if USE_TDM
        // ---- A tile via Tensor Data Mover (one wave issues the DMA) ----
        if (wave == 0) {
            const unsigned long long ga = wbase + (unsigned long long)k0 * 2ull;
            const int remK = Ktot - k0;       // HW OOB zero-fill handles edges
            const int remM = Cout - m0;
            u32x4 g0;
            g0[0] = 1u;                                   // count=1 (valid D#)
            g0[1] = ldsA;                                 // lds_addr
            g0[2] = (unsigned)(ga & 0xffffffffull);       // global_addr[31:0]
            g0[3] = (unsigned)((ga >> 32) & 0x01ffffffull) | (2u << 30); // type=2
            i32x8 g1;
            g1[0] = (1 << 16)      // data_size = 2 bytes
                  | (1 << 20)      // pad_enable
                  | (3 << 22)      // pad_interval: every 16 dwords (=32 halfs)
                  | (3 << 25);     // pad_amount: 4 dwords (=8 halfs) -> stride 40
            g1[1] = (remK & 0xffff) << 16;                       // tensor_dim0 lo
            g1[2] = ((remK >> 16) & 0xffff) | ((remM & 0xffff) << 16); // dim0 hi, dim1 lo
            g1[3] = (32 << 16);                                  // tile_dim0 = 32
            g1[4] = MTILE;                                       // tile_dim1
            g1[5] = Ktot;                                        // dim0_stride lo
            g1[6] = 0;
            g1[7] = 0;
            const i32x4 gz4 = (i32x4){0, 0, 0, 0};               // 2D: groups 2/3 null
            const i32x8 gz8 = (i32x8){0, 0, 0, 0, 0, 0, 0, 0};
            __builtin_amdgcn_tensor_load_to_lds(g0, g1, gz4, gz4, gz8, 0);
        }
#else
        // ---- A tile staged by VALU from packed f16 weights ----
        {
            constexpr int APT = MW * 4;
            const int base = tid * APT;
            const int row  = base >> 5;
            const int kk0  = base & 31;
            const int m    = m0 + row;
            #pragma unroll
            for (int i = 0; i < APT; ++i) {
                const int k = k0 + kk0 + i;
                _Float16 v = (_Float16)0.f;
                if (m < Cout && k < Ktot) v = W[(long)m * Ktot + k];
                sA[row * LDSW + kk0 + i] = v;
            }
        }
#endif
        // ---- stage B (im2col gather) into col-major LDS, overlaps the DMA ----
        for (int c = 0; c < nCols; ++c) {
            const long base = cbase[c];
            const int  y0 = cy0[c], x0 = cx0[c];
            const int  okc = cok[c];
            for (int kk = kkBeg; kk < kkEnd; ++kk) {
                const int k = k0 + kk;
                float v = 0.f;
                if (okc && k < Ktot) {
                    const int ci = k / (KH * KW);
                    const int r  = k - ci * (KH * KW);
                    const int kh = r / KW, kw = r - kh * KW;
                    if (TR) { // stride-2 transposed conv: oy = 2*iy - P + kh
                        const int ty = y0 - kh, tx = x0 - kw;
                        if (ty >= 0 && tx >= 0 && !(ty & 1) && !(tx & 1)) {
                            const int iy = ty >> 1, ix = tx >> 1;
                            if (iy < Hin && ix < Win)
                                v = X[base + ((long)ci * Hin + iy) * Win + ix];
                        }
                    } else {
                        const int iy = y0 + kh, ix = x0 + kw;
                        if (iy >= 0 && iy < Hin && ix >= 0 && ix < Win)
                            v = X[base + ((long)ci * Hin + iy) * Win + ix];
                    }
                    if (PRE) v = v > 0.f ? v : 0.f;   // fused input ReLU
                }
                sB[myCol[c] * LDSW + kk] = (_Float16)v;
            }
        }
#if USE_TDM
        if (wave == 0) __builtin_amdgcn_s_wait_tensorcnt(0);
#endif
        __syncthreads();

        // ---- fragments per ISA 7.12.2 layouts; 1 A reused over 4 WMMAs ----
        const int hsel = lane >> 4;          // 0: lanes 0-15, 1: lanes 16-31
        const int r16  = lane & 15;
        const int mgrp = (MW == 1) ? 0 : (wave & (MW - 1));
        const int ngrp = (MW == 4) ? 0 : (wave >> (MW == 2 ? 1 : 0));
        const int arow = mgrp * 16 + r16;
        // A lane: low lanes K[0..7]+[16..23]; high lanes K[8..15]+[24..31]
        const v8h alo = *(const v8h*)&sA[arow * LDSW + hsel * 8];
        const v8h ahi = *(const v8h*)&sA[arow * LDSW + hsel * 8 + 16];
        v16h a;
        #pragma unroll
        for (int e = 0; e < 8; ++e) { a[e] = alo[e]; a[e + 8] = ahi[e]; }
        #pragma unroll
        for (int rep = 0; rep < 4; ++rep) {
            // B lane: col fixed, low lanes K[0..7]+[16..23]; high lanes K[8..15]+[24..31]
            const int col = ngrp * 64 + rep * 16 + r16;
            const v8h blo = *(const v8h*)&sB[col * LDSW + hsel * 8];
            const v8h bhi = *(const v8h*)&sB[col * LDSW + hsel * 8 + 16];
            v16h b;
            #pragma unroll
            for (int e = 0; e < 8; ++e) { b[e] = blo[e]; b[e + 8] = bhi[e]; }
            acc[rep] = __builtin_amdgcn_wmma_f32_16x16x32_f16(
                           false, a, false, b, (short)0, acc[rep], false, false);
        }
    }

    // ---- store D (16x16 f32): VGPR v -> row mbase+v, col = lane&15 ----
    const int r16   = lane & 15;
    const int mbase = (lane >> 4) * 8;
    const int mgrp  = (MW == 1) ? 0 : (wave & (MW - 1));
    const int ngrp  = (MW == 4) ? 0 : (wave >> (MW == 2 ? 1 : 0));
    const int mrow0 = m0 + mgrp * 16 + mbase;
    #pragma unroll
    for (int rep = 0; rep < 4; ++rep) {
        const int n = n0 + ngrp * 64 + rep * 16 + r16;
        if (n >= Ntot) continue;
        if (VQ) {
            // distances[n, m] = ||x_n||^2 + ||e_m||^2 - 2 <x_n, e_m>
            const float xn = aux[n];
            #pragma unroll
            for (int v = 0; v < 8; ++v) {
                const int m = mrow0 + v;
                if (m < Cout) out[(long)n * Cout + m] = xn + bias[m] - 2.f * acc[rep][v];
            }
        } else {
            const int ox = n % Wout; int t = n / Wout;
            const int oy = t % Hout; const int b = t / Hout;
            #pragma unroll
            for (int v = 0; v < 8; ++v) {
                const int m = mrow0 + v;
                if (m < Cout) {
                    float val = acc[rep][v];
                    if (bias) val += bias[m];
                    const long oidx = ((long)(b * Cout + m) * Hout + oy) * Wout + ox;
                    if (RES)  val += resid[oidx];
                    if (POST) val = val > 0.f ? val : 0.f;
                    out[oidx] = val;
                }
            }
        }
    }
}

// ------------------------------ helper kernels ------------------------------
// Pack weights to f16 (Cout x K) row-major; TR also folds the
// (Cin,Cout,kh,kw) -> (Cout, ci*KK+r) transpose for ConvTranspose layers.
__global__ void pack_w(const float* __restrict__ src, _Float16* __restrict__ dst,
                       int Cout, int K, int KK, int TR) {
    const int i = blockIdx.x * blockDim.x + threadIdx.x;
    if (i >= Cout * K) return;
    const int m = i / K, k = i - m * K;
    float v;
    if (TR) { const int ci = k / KK, r = k - ci * KK;
              v = src[((long)ci * Cout + m) * KK + r]; }
    else      v = src[i];
    dst[i] = (_Float16)v;
}

__global__ void zero_f32(float* p, long n) {
    long i = (long)blockIdx.x * blockDim.x + threadIdx.x;
    const long st = (long)gridDim.x * blockDim.x;
    for (; i < n; i += st) p[i] = 0.f;
}

__global__ void rowsq_pix(const float* __restrict__ z, float* __restrict__ xsq,
                          int Bn, int D, int H, int Wd) {
    const int n = blockIdx.x * blockDim.x + threadIdx.x;
    const int Nt = Bn * H * Wd;
    if (n >= Nt) return;
    const int x = n % Wd; int t = n / Wd;
    const int y = t % H;  const int b = t / H;
    float s = 0.f;
    for (int d = 0; d < D; ++d) {
        const float v = z[((long)(b * D + d) * H + y) * Wd + x];
        s += v * v;
    }
    xsq[n] = s;
}

__global__ void codesq(const float* __restrict__ E, float* __restrict__ esq,
                       int K, int D) {
    const int k = blockIdx.x * blockDim.x + threadIdx.x;
    if (k >= K) return;
    float s = 0.f;
    for (int d = 0; d < D; ++d) { const float v = E[(long)k * D + d]; s += v * v; }
    esq[k] = s;
}

__global__ void argmin_vq(const float* __restrict__ dist, float* __restrict__ idx_f,
                          int* __restrict__ idx_i, float* __restrict__ enc,
                          unsigned int* __restrict__ counts, int Nt, int K) {
    const int n = blockIdx.x * blockDim.x + threadIdx.x;
    if (n >= Nt) return;
    const float* row = dist + (long)n * K;
    float best = row[0]; int bi = 0;
    for (int k = 1; k < K; ++k) { const float v = row[k]; if (v < best) { best = v; bi = k; } }
    idx_f[n] = (float)bi;
    idx_i[n] = bi;
    enc[(long)n * K + bi] = 1.0f;
    atomicAdd(&counts[bi], 1u);
}

__global__ void quantize_nchw(const float* __restrict__ E, const int* __restrict__ idx,
                              float* __restrict__ qz, int Bn, int D, int H, int Wd) {
    const long i = (long)blockIdx.x * blockDim.x + threadIdx.x;
    const long tot = (long)Bn * D * H * Wd;
    if (i >= tot) return;
    const int x = (int)(i % Wd); long t = i / Wd;
    const int y = (int)(t % H);  t /= H;
    const int d = (int)(t % D);  const int b = (int)(t / D);
    const int n = (b * H + y) * Wd + x;
    qz[i] = E[(long)idx[n] * D + d];
}

__global__ void mse_loss(const float* __restrict__ a, const float* __restrict__ b,
                         float* __restrict__ loss, long n, float scale) {
    __shared__ float red[256];
    long i = (long)blockIdx.x * blockDim.x + threadIdx.x;
    const long st = (long)gridDim.x * blockDim.x;
    float s = 0.f;
    for (; i < n; i += st) { const float d = a[i] - b[i]; s += d * d; }
    red[threadIdx.x] = s;
    __syncthreads();
    for (int o = 128; o > 0; o >>= 1) {
        if ((int)threadIdx.x < o) red[threadIdx.x] += red[threadIdx.x + o];
        __syncthreads();
    }
    if (threadIdx.x == 0) atomicAdd(loss, red[0] * scale);
}

__global__ void perplexity_k(const unsigned int* __restrict__ counts,
                             float* __restrict__ out, int K, float invN) {
    __shared__ float red[512];
    const int k = threadIdx.x;
    const float p = (k < K) ? counts[k] * invN : 0.f;
    red[k] = p * logf(p + 1e-10f);
    __syncthreads();
    for (int o = 256; o > 0; o >>= 1) {
        if (k < o) red[k] += red[k + o];
        __syncthreads();
    }
    if (k == 0) out[0] = expf(-red[0]);
}

// ------------------------------- launcher -----------------------------------
static inline dim3 cgrid(int Ntot, int Cout, int MW) {
    const int NC = (4 / MW) * 64;
    const int MT = MW * 16;
    return dim3((unsigned)((Ntot + NC - 1) / NC), (unsigned)((Cout + MT - 1) / MT));
}

extern "C" void kernel_launch(void* const* d_in, const int* in_sizes, int n_in,
                              void* d_out, int out_size, void* d_ws, size_t ws_size,
                              hipStream_t stream) {
    (void)in_sizes; (void)n_in; (void)out_size; (void)ws_size;
    // inputs in setup_inputs order
    const float* x     = (const float*)d_in[0];
    const float* e_w1  = (const float*)d_in[1];  const float* e_b1 = (const float*)d_in[2];
    const float* e_w2  = (const float*)d_in[3];  const float* e_b2 = (const float*)d_in[4];
    const float* e_w3  = (const float*)d_in[5];  const float* e_b3 = (const float*)d_in[6];
    const float* er0w1 = (const float*)d_in[7];  const float* er0w2 = (const float*)d_in[8];
    const float* er1w1 = (const float*)d_in[9];  const float* er1w2 = (const float*)d_in[10];
    const float* p_w   = (const float*)d_in[11]; const float* p_b  = (const float*)d_in[12];
    const float* E     = (const float*)d_in[13];
    const float* d_w1  = (const float*)d_in[14]; const float* d_b1 = (const float*)d_in[15];
    const float* dr0w1 = (const float*)d_in[16]; const float* dr0w2 = (const float*)d_in[17];
    const float* dr1w1 = (const float*)d_in[18]; const float* dr1w2 = (const float*)d_in[19];
    const float* dt_w1 = (const float*)d_in[20]; const float* dt_b1 = (const float*)d_in[21];
    const float* dt_w2 = (const float*)d_in[22]; const float* dt_b2 = (const float*)d_in[23];

    float* out = (float*)d_out;
    // output layout: loss | x_recon | perplexity | distances | idx | encodings
    const long LOSS = 0;
    const long XREC = 1;
    const long PERP = XREC + (long)16 * 3 * 256 * 256;          // 3145729
    const long DIST = PERP + 1;                                 // 3145730
    const long IDXO = DIST + (long)65536 * 512;                 // 36700162
    const long ENCO = IDXO + 65536;                             // 36765698

    // workspace carve-up (~178 MB)
    float* ws = (float*)d_ws;
    long o = 0;
    float* z1   = ws + o; o += (long)16 * 64 * 128 * 128;   // 16.8M  (reused by dt1 out)
    float* z2   = ws + o; o += (long)16 * 128 * 64 * 64;    // 8.4M   (reused by decoder y1)
    float* z3   = ws + o; o += (long)16 * 128 * 64 * 64;    // 8.4M
    float* rtmp = ws + o; o += (long)16 * 32 * 64 * 64;     // 2.1M
    float* ze   = ws + o; o += (long)16 * 64 * 64 * 64;     // 4.2M
    float* qz   = ws + o; o += (long)16 * 64 * 64 * 64;     // 4.2M
    float* xsq  = ws + o; o += 65536;
    float* esq  = ws + o; o += 512;
    unsigned int* counts = (unsigned int*)(ws + o); o += 512;
    int* idxI = (int*)(ws + o); o += 65536;
    // packed f16 weights (Cout x Ktot each), 16-half aligned
    _Float16* hw = (_Float16*)(ws + o);
    long ho = 0;
    auto nexth = [&](long n) { _Float16* p = hw + ho; ho += (n + 15) & ~15L; return p; };
    _Float16* wEn1 = nexth(64 * 48);     _Float16* wEn2 = nexth(128 * 1024);
    _Float16* wEn3 = nexth(128 * 1152);
    _Float16* wR0a = nexth(32 * 1152);   _Float16* wR0b = nexth(128 * 32);
    _Float16* wR1a = nexth(32 * 1152);   _Float16* wR1b = nexth(128 * 32);
    _Float16* wP   = nexth(64 * 128);    _Float16* wE   = nexth(512 * 64);
    _Float16* wD1  = nexth(128 * 576);
    _Float16* wS0a = nexth(32 * 1152);   _Float16* wS0b = nexth(128 * 32);
    _Float16* wS1a = nexth(32 * 1152);   _Float16* wS1b = nexth(128 * 32);
    _Float16* wT1  = nexth(64 * 2048);   _Float16* wT2  = nexth(3 * 1024);

    const dim3 blk(128);
    auto pk = [&](const float* s, _Float16* d, int Co, int K, int KK, int TR) {
        pack_w<<<dim3((unsigned)((Co * K + 255) / 256)), dim3(256), 0, stream>>>(s, d, Co, K, KK, TR);
    };

    // -- pack all weights to f16 GEMM layout (feeds the TDM A-tile loads) --
    pk(e_w1, wEn1, 64, 48, 16, 0);   pk(e_w2, wEn2, 128, 1024, 16, 0);
    pk(e_w3, wEn3, 128, 1152, 9, 0);
    pk(er0w1, wR0a, 32, 1152, 9, 0); pk(er0w2, wR0b, 128, 32, 1, 0);
    pk(er1w1, wR1a, 32, 1152, 9, 0); pk(er1w2, wR1b, 128, 32, 1, 0);
    pk(p_w, wP, 64, 128, 1, 0);      pk(E, wE, 512, 64, 1, 0);
    pk(d_w1, wD1, 128, 576, 9, 0);
    pk(dr0w1, wS0a, 32, 1152, 9, 0); pk(dr0w2, wS0b, 128, 32, 1, 0);
    pk(dr1w1, wS1a, 32, 1152, 9, 0); pk(dr1w2, wS1b, 128, 32, 1, 0);
    pk(dt_w1, wT1, 64, 2048, 16, 1); pk(dt_w2, wT2, 3, 1024, 16, 1);

    // -- init output accumulators --
    zero_f32<<<2048, 256, 0, stream>>>(out + ENCO, (long)65536 * 512);
    zero_f32<<<1, 256, 0, stream>>>((float*)counts, 512);
    zero_f32<<<1, 64, 0, stream>>>(out + LOSS, 1);

    // -- encoder --
    conv_wmma<4,4,2,1, 0,1,0,0,0, 4><<<cgrid(16*128*128, 64, 4), blk, 0, stream>>>(
        x, wEn1, e_b1, nullptr, nullptr, z1, 16, 3, 64, 256, 256, 128, 128);
    conv_wmma<4,4,2,1, 0,1,0,0,0, 4><<<cgrid(16*64*64, 128, 4), blk, 0, stream>>>(
        z1, wEn2, e_b2, nullptr, nullptr, z2, 16, 64, 128, 128, 128, 64, 64);
    conv_wmma<3,3,1,1, 0,0,0,0,0, 4><<<cgrid(16*64*64, 128, 4), blk, 0, stream>>>(
        z2, wEn3, e_b3, nullptr, nullptr, z3, 16, 128, 128, 64, 64, 64, 64);
    // encoder residual stack (ReLU->3x3->ReLU->1x1, residual add), x2
    conv_wmma<3,3,1,1, 1,1,0,0,0, 2><<<cgrid(16*64*64, 32, 2), blk, 0, stream>>>(
        z3, wR0a, nullptr, nullptr, nullptr, rtmp, 16, 128, 32, 64, 64, 64, 64);
    conv_wmma<1,1,1,0, 0,0,1,0,0, 4><<<cgrid(16*64*64, 128, 4), blk, 0, stream>>>(
        rtmp, wR0b, nullptr, z3, nullptr, z3, 16, 32, 128, 64, 64, 64, 64);
    conv_wmma<3,3,1,1, 1,1,0,0,0, 2><<<cgrid(16*64*64, 32, 2), blk, 0, stream>>>(
        z3, wR1a, nullptr, nullptr, nullptr, rtmp, 16, 128, 32, 64, 64, 64, 64);
    conv_wmma<1,1,1,0, 0,0,1,0,0, 4><<<cgrid(16*64*64, 128, 4), blk, 0, stream>>>(
        rtmp, wR1b, nullptr, z3, nullptr, z3, 16, 32, 128, 64, 64, 64, 64);
    // pre-VQ 1x1 (stack's trailing ReLU fused as PRE)
    conv_wmma<1,1,1,0, 1,0,0,0,0, 4><<<cgrid(16*64*64, 64, 4), blk, 0, stream>>>(
        z3, wP, p_b, nullptr, nullptr, ze, 16, 128, 64, 64, 64, 64, 64);

    // -- vector quantizer --
    rowsq_pix<<<256, 256, 0, stream>>>(ze, xsq, 16, 64, 64, 64);
    codesq<<<2, 256, 0, stream>>>(E, esq, 512, 64);
    // distances = xsq + esq - 2 * (ze . E^T): WMMA GEMM with E as 512x64 weight
    conv_wmma<1,1,1,0, 0,0,0,0,1, 4><<<cgrid(16*64*64, 512, 4), blk, 0, stream>>>(
        ze, wE, esq, nullptr, xsq, out + DIST, 16, 64, 512, 64, 64, 64, 64);
    argmin_vq<<<256, 256, 0, stream>>>(out + DIST, out + IDXO, idxI, out + ENCO,
                                       counts, 65536, 512);
    quantize_nchw<<<16384, 256, 0, stream>>>(E, idxI, qz, 16, 64, 64, 64);
    // loss = q_latent + 0.25*e_latent = 1.25 * mean((q - z)^2)  (forward values equal)
    mse_loss<<<2048, 256, 0, stream>>>(qz, ze, out + LOSS, (long)16*64*64*64,
                                       1.25f / (float)(16*64*64*64));
    perplexity_k<<<1, 512, 0, stream>>>(counts, out + PERP, 512, 1.f / 65536.f);

    // -- decoder (straight-through forward value == quantized) --
    conv_wmma<3,3,1,1, 0,0,0,0,0, 4><<<cgrid(16*64*64, 128, 4), blk, 0, stream>>>(
        qz, wD1, d_b1, nullptr, nullptr, z2, 16, 64, 128, 64, 64, 64, 64);
    conv_wmma<3,3,1,1, 1,1,0,0,0, 2><<<cgrid(16*64*64, 32, 2), blk, 0, stream>>>(
        z2, wS0a, nullptr, nullptr, nullptr, rtmp, 16, 128, 32, 64, 64, 64, 64);
    conv_wmma<1,1,1,0, 0,0,1,0,0, 4><<<cgrid(16*64*64, 128, 4), blk, 0, stream>>>(
        rtmp, wS0b, nullptr, z2, nullptr, z2, 16, 32, 128, 64, 64, 64, 64);
    conv_wmma<3,3,1,1, 1,1,0,0,0, 2><<<cgrid(16*64*64, 32, 2), blk, 0, stream>>>(
        z2, wS1a, nullptr, nullptr, nullptr, rtmp, 16, 128, 32, 64, 64, 64, 64);
    conv_wmma<1,1,1,0, 0,0,1,0,0, 4><<<cgrid(16*64*64, 128, 4), blk, 0, stream>>>(
        rtmp, wS1b, nullptr, z2, nullptr, z2, 16, 32, 128, 64, 64, 64, 64);
    // transposed convs (stride 2, pad 1); stack's trailing ReLU fused as PRE on dt1
    conv_wmma<4,4,2,1, 1,1,0,1,0, 4><<<cgrid(16*128*128, 64, 4), blk, 0, stream>>>(
        z2, wT1, dt_b1, nullptr, nullptr, z1, 16, 128, 64, 64, 64, 128, 128);
    conv_wmma<4,4,2,1, 0,0,0,1,0, 1><<<cgrid(16*256*256, 3, 1), blk, 0, stream>>>(
        z1, wT2, dt_b2, nullptr, nullptr, out + XREC, 16, 64, 3, 128, 128, 256, 256);
}
